// ParallelNestedPerceiverEncoder_38800734552707
// MI455X (gfx1250) — compile-verified, matrix operand order checked by reading, not run on hardware
//
#include <hip/hip_runtime.h>
#include <hip/hip_bf16.h>

// ---------------------------------------------------------------------------
// Model constants (from the reference)
// ---------------------------------------------------------------------------
#define MB   32      // batch
#define NC   2048    // context tokens
#define NT   2048    // target tokens
#define LQ   128     // latents
#define DM   256     // model dim
#define NH   8       // heads
#define DH   32      // head dim  (== WMMA bf16 K!)
#define DFF  1024
#define NLAYERS 2

using bf16 = __bf16;
typedef __attribute__((ext_vector_type(16))) __bf16 v16bf;
typedef __attribute__((ext_vector_type(8)))  __bf16 v8bf;
typedef __attribute__((ext_vector_type(8)))  float  v8f;
typedef __attribute__((ext_vector_type(4)))  float  v4f;

// ---------------------------------------------------------------------------
// WMMA helpers
// ---------------------------------------------------------------------------
__device__ inline v8f wmma_bf16(v16bf a, v16bf b, v8f c) {
  // D = A(16x32) * B(32x16) + C(16x16), f32 accumulate
  return __builtin_amdgcn_wmma_f32_16x16x32_bf16(
      /*neg_a=*/false, a, /*neg_b=*/false, b,
      /*c_mod=*/(short)0, c, /*reuse_a=*/false, /*reuse_b=*/false);
}

// A-fragment loader (row-major source, 16 rows x 32 cols, bf16).
// CDNA5 16-bit A layout: lane = row + 16*half; per lane two contiguous runs
// of 8 bf16: K = 8*half + [0..7]  and  K = 16 + 8*half + [0..7].
// B-fragments are loaded as A-fragments of B^T (weights/V are pre-transposed).
__device__ inline v16bf frag_ld(const bf16* __restrict__ p0, int ld) {
  int lane = threadIdx.x & 31;
  int half = lane >> 4;
  int row  = lane & 15;
  const bf16* p = p0 + row * ld + (half << 3);
  v8bf lo = *(const v8bf*)(p);
  v8bf hi = *(const v8bf*)(p + 16);
  return __builtin_shufflevector(lo, hi, 0,1,2,3,4,5,6,7,8,9,10,11,12,13,14,15);
}

// 16-lane group max reduction (C-matrix rows live across 16 lanes)
__device__ inline float redmax16(float v) {
#pragma unroll
  for (int m = 1; m < 16; m <<= 1) v = fmaxf(v, __shfl_xor(v, m, 32));
  return v;
}

// gelu(tanh approx) via sigmoid: 0.5*(1+tanh(z)) == sigmoid(2z); lowers to
// v_exp_f32 + v_rcp instead of a libm tanh polynomial.
__device__ inline float gelu_tanh(float x) {
  float y = 1.5957691216057308f * (x + 0.044715f * x * x * x);
  return x / (1.0f + __expf(-y));
}

// ---------------------------------------------------------------------------
// Weight prep: Wt[n*K + k] = bf16(W[k*N + n])   (W is KxN row-major)
// ---------------------------------------------------------------------------
__global__ __launch_bounds__(256) void wprep_kernel(const float* __restrict__ W,
                                                    bf16* __restrict__ Wt,
                                                    int K, int N) {
  int idx = blockIdx.x * 256 + threadIdx.x;
  if (idx >= K * N) return;
  int n = idx / K;
  int k = idx - n * K;
  Wt[(size_t)n * K + k] = (bf16)W[(size_t)k * N + n];
}

// broadcast latents -> xq (32 copies)
__global__ __launch_bounds__(256) void bcast_kernel(const float* __restrict__ lat,
                                                    float* __restrict__ xq) {
  int idx = blockIdx.x * 256 + threadIdx.x;   // grid covers 32*128*256
  xq[idx] = lat[idx & (LQ * DM - 1)];
}

// ---------------------------------------------------------------------------
// LayerNorm (D=256): one wave per row, writes bf16
// ---------------------------------------------------------------------------
__global__ __launch_bounds__(256) void ln_kernel(const float* __restrict__ x,
                                                 const float* __restrict__ g,
                                                 const float* __restrict__ b,
                                                 bf16* __restrict__ y) {
  int w    = threadIdx.x >> 5;
  int lane = threadIdx.x & 31;
  size_t row = (size_t)blockIdx.x * 8 + w;
  const float* xr = x + row * DM + lane * 8;
  v4f a = *(const v4f*)(xr);
  v4f c = *(const v4f*)(xr + 4);
  float s = 0.f, ss = 0.f;
#pragma unroll
  for (int i = 0; i < 4; i++) {
    s  += a[i] + c[i];
    ss += a[i] * a[i] + c[i] * c[i];
  }
#pragma unroll
  for (int m = 1; m < 32; m <<= 1) { s += __shfl_xor(s, m, 32); ss += __shfl_xor(ss, m, 32); }
  float mu  = s * (1.0f / DM);
  float var = ss * (1.0f / DM) - mu * mu;
  float rs  = rsqrtf(var + 1e-5f);
  const float* gg = g + lane * 8;
  const float* bb = b + lane * 8;
  v8bf o;
#pragma unroll
  for (int i = 0; i < 4; i++) {
    o[i]     = (bf16)((a[i] - mu) * rs * gg[i]     + bb[i]);
    o[4 + i] = (bf16)((c[i] - mu) * rs * gg[4 + i] + bb[4 + i]);
  }
  *(v8bf*)(y + row * DM + lane * 8) = o;
}

// ---------------------------------------------------------------------------
// GEMM: C[M,N] = A[M,K](bf16) @ W[K,N], W given pre-transposed as Wt[N,K] bf16.
// LDS-staged + double-buffered: block tile 128x128, K-step 32, 256 threads =
// 8 waves, each wave a 32x64 tile (2x4 WMMA accumulators). ~64 flops per L2
// byte. LDS rows padded to 40 bf16 (80B): 16B-aligned b128 and conflict-free
// fragment reads (r*20 mod 64 is a permutation over the 16 lanes).
// OP 0: store bf16 * cscale    OP 1: gelu -> bf16
// OP 2: Cf = Cin + acc (f32)   OP 3: store bf16 transposed per head into
//                                    vt[m][h][dh][nk]  (nk = 1<<nkshift)
// ---------------------------------------------------------------------------
#define LDP 40
template <int OP>
__global__ __launch_bounds__(256) void gemm_kernel(const bf16* __restrict__ A,
                                                   const bf16* __restrict__ Wt,
                                                   const float* __restrict__ Cin,
                                                   float* __restrict__ Cf,
                                                   bf16* __restrict__ Cb,
                                                   int M, int N, int K,
                                                   int nkshift, float cscale) {
  __shared__ bf16 As[2][128 * LDP];
  __shared__ bf16 Bs[2][128 * LDP];

  int w  = threadIdx.x >> 5;
  int wm = w & 3;           // 4 waves along M
  int wn = w >> 2;          // 2 waves along N
  int rowbase = blockIdx.y * 128 + wm * 32;
  int colbase = blockIdx.x * 128 + wn * 64;

  // cooperative fill mapping: thread -> (row in tile, 16-elem k chunk)
  int lr = threadIdx.x >> 1;
  int lk = (threadIdx.x & 1) << 4;
  const bf16* aptr = A  + (size_t)(blockIdx.y * 128 + lr) * K + lk;
  const bf16* bptr = Wt + (size_t)(blockIdx.x * 128 + lr) * K + lk;

  v8f acc[2][4];
#pragma unroll
  for (int i = 0; i < 2; i++)
#pragma unroll
    for (int j = 0; j < 4; j++) acc[i][j] = {};

  // prefetch k0 = 0
  v8bf ax = ((const v8bf*)aptr)[0], ay = ((const v8bf*)aptr)[1];
  v8bf bx = ((const v8bf*)bptr)[0], by = ((const v8bf*)bptr)[1];
  int cur = 0;
  {
    *(v8bf*)&As[0][lr * LDP + lk]     = ax;
    *(v8bf*)&As[0][lr * LDP + lk + 8] = ay;
    *(v8bf*)&Bs[0][lr * LDP + lk]     = bx;
    *(v8bf*)&Bs[0][lr * LDP + lk + 8] = by;
  }

  for (int k0 = 0; k0 < K; k0 += 32) {
    __syncthreads();
    bool more = (k0 + 32) < K;
    if (more) {
      ax = ((const v8bf*)(aptr + k0 + 32))[0];
      ay = ((const v8bf*)(aptr + k0 + 32))[1];
      bx = ((const v8bf*)(bptr + k0 + 32))[0];
      by = ((const v8bf*)(bptr + k0 + 32))[1];
    }
    v16bf a0 = frag_ld(&As[cur][(wm * 32 +  0) * LDP], LDP);
    v16bf a1 = frag_ld(&As[cur][(wm * 32 + 16) * LDP], LDP);
    v16bf b0 = frag_ld(&Bs[cur][(wn * 64 +  0) * LDP], LDP);
    v16bf b1 = frag_ld(&Bs[cur][(wn * 64 + 16) * LDP], LDP);
    v16bf b2 = frag_ld(&Bs[cur][(wn * 64 + 32) * LDP], LDP);
    v16bf b3 = frag_ld(&Bs[cur][(wn * 64 + 48) * LDP], LDP);
    acc[0][0] = wmma_bf16(a0, b0, acc[0][0]);
    acc[0][1] = wmma_bf16(a0, b1, acc[0][1]);
    acc[0][2] = wmma_bf16(a0, b2, acc[0][2]);
    acc[0][3] = wmma_bf16(a0, b3, acc[0][3]);
    acc[1][0] = wmma_bf16(a1, b0, acc[1][0]);
    acc[1][1] = wmma_bf16(a1, b1, acc[1][1]);
    acc[1][2] = wmma_bf16(a1, b2, acc[1][2]);
    acc[1][3] = wmma_bf16(a1, b3, acc[1][3]);
    if (more) {
      int nxt = cur ^ 1;
      *(v8bf*)&As[nxt][lr * LDP + lk]     = ax;
      *(v8bf*)&As[nxt][lr * LDP + lk + 8] = ay;
      *(v8bf*)&Bs[nxt][lr * LDP + lk]     = bx;
      *(v8bf*)&Bs[nxt][lr * LDP + lk + 8] = by;
      cur = nxt;
    }
  }

  int lane  = threadIdx.x & 31;
  int half  = lane >> 4;
  int col16 = lane & 15;
  int nkmask = (1 << nkshift) - 1;
#pragma unroll
  for (int i = 0; i < 2; i++) {
#pragma unroll
    for (int j = 0; j < 4; j++) {
#pragma unroll
      for (int v = 0; v < 8; v++) {
        int r = rowbase + i * 16 + v + half * 8;
        int c = colbase + j * 16 + col16;
        float val = acc[i][j][v];
        if (OP == 0) {
          Cb[(size_t)r * N + c] = (bf16)(val * cscale);
        } else if (OP == 1) {
          Cb[(size_t)r * N + c] = (bf16)gelu_tanh(val);
        } else if (OP == 2) {
          Cf[(size_t)r * N + c] = Cin[(size_t)r * N + c] + val;
        } else {
          // transposed-per-head V store: vt[((m*NH + h)*DH + dh)][token]
          int m  = r >> nkshift;
          int tk = r & nkmask;
          int h  = c >> 5;
          int dh = c & 31;
          size_t idx = (((size_t)(m * NH + h) * DH + dh) << nkshift) + tk;
          Cb[idx] = (bf16)val;
        }
      }
    }
  }
}

// ---------------------------------------------------------------------------
// Fused multi-head attention, flash-style. One wave per (m, h, 32-query pair
// of 16-row tiles): K/V fragments are shared across the two query tiles,
// halving K/V L2 traffic. DH = 32 == WMMA K, so scores = one WMMA per tile.
// V is pre-transposed per head (GEMM OP 3) -> V B-frags are direct global
// b128 fragment loads. Row sums of P come from one extra WMMA against an
// all-ones B matrix (layout-proof), removing the sum butterfly entirely;
// only the row-max butterfly (needed before exp) remains.
// Q arrives pre-scaled by 1/sqrt(DH) from its projection GEMM.
// ---------------------------------------------------------------------------
__global__ __launch_bounds__(256) void attn_kernel(const bf16* __restrict__ qb,
                                                   const bf16* __restrict__ kb,
                                                   const bf16* __restrict__ vt,
                                                   bf16* __restrict__ ob,
                                                   int nq, int nk) {
  __shared__ bf16 Pb[8][2 * 16 * 32];     // 16 KB

  int w     = threadIdx.x >> 5;
  int lane  = threadIdx.x & 31;
  int half  = lane >> 4;
  int col16 = lane & 15;

  int nqt  = nq >> 5;                  // 32-query groups
  int task = blockIdx.x * 8 + w;
  int qt   = task % nqt;
  int rest = task / nqt;
  int h    = rest & 7;
  int m    = rest >> 3;

  const bf16* kbase = kb + ((size_t)m * nk) * DM + h * DH;
  const bf16* vbase = vt + ((size_t)(m * NH + h) * DH) * nk;

  v16bf qf[2];
  qf[0] = frag_ld(qb + ((size_t)(m * nq + qt * 32))      * DM + h * DH, DM);
  qf[1] = frag_ld(qb + ((size_t)(m * nq + qt * 32 + 16)) * DM + h * DH, DM);

  v16bf ones;
#pragma unroll
  for (int i = 0; i < 16; i++) ones[i] = (bf16)1.0f;

  v8f o[2][2];
  float mprev[2][8], lsum[2][8];
#pragma unroll
  for (int t = 0; t < 2; t++) {
    o[t][0] = {}; o[t][1] = {};
#pragma unroll
    for (int j = 0; j < 8; j++) { mprev[t][j] = -3.0e38f; lsum[t][j] = 0.0f; }
  }

  for (int kt = 0; kt < nk; kt += 32) {
    // ---- shared K / V fragments for both query tiles ----
    v16bf bk0 = frag_ld(kbase + (size_t)kt * DM,        DM);
    v16bf bk1 = frag_ld(kbase + (size_t)(kt + 16) * DM, DM);
    v16bf bv0 = frag_ld(vbase + kt,           nk);   // dh 0..15
    v16bf bv1 = frag_ld(vbase + 16 * nk + kt, nk);   // dh 16..31

    float corr[2][8];
#pragma unroll
    for (int t = 0; t < 2; t++) {
      v8f z = {};
      v8f s0 = wmma_bf16(qf[t], bk0, z);   // keys kt..kt+15 (pre-scaled Q)
      v8f s1 = wmma_bf16(qf[t], bk1, z);   // keys kt+16..kt+31
#pragma unroll
      for (int j = 0; j < 8; j++) {
        float a0 = s0[j], a1 = s1[j];
        float tm = redmax16(fmaxf(a0, a1));
        float mn = fmaxf(mprev[t][j], tm);
        float cj = __expf(mprev[t][j] - mn);
        corr[t][j]  = cj;
        mprev[t][j] = mn;
        float p0 = __expf(a0 - mn);
        float p1 = __expf(a1 - mn);
        o[t][0][j] *= cj;
        o[t][1][j] *= cj;
        int rl = j + half * 8;
        Pb[w][t * 512 + rl * 32 + col16]      = (bf16)p0;
        Pb[w][t * 512 + rl * 32 + 16 + col16] = (bf16)p1;
      }
    }
    __syncthreads();

    // ---- O += P*V ; row-sums via P*Ones (one WMMA, no butterflies) ----
#pragma unroll
    for (int t = 0; t < 2; t++) {
      v16bf pf = frag_ld(&Pb[w][t * 512], 32);
      o[t][0] = wmma_bf16(pf, bv0, o[t][0]);
      o[t][1] = wmma_bf16(pf, bv1, o[t][1]);
      v8f z = {};
      v8f srow = wmma_bf16(pf, ones, z);
#pragma unroll
      for (int j = 0; j < 8; j++) lsum[t][j] = lsum[t][j] * corr[t][j] + srow[j];
    }
    __syncthreads();
  }

  // ---- normalize and write bf16 output (consumed by the Wo GEMM) ----
#pragma unroll
  for (int t = 0; t < 2; t++) {
#pragma unroll
    for (int j = 0; j < 8; j++) {
      float inv = 1.0f / lsum[t][j];
      size_t r = (size_t)(m * nq + qt * 32 + t * 16 + j + half * 8) * DM + h * DH;
      ob[r + col16]      = (bf16)(o[t][0][j] * inv);
      ob[r + 16 + col16] = (bf16)(o[t][1][j] * inv);
    }
  }
}

// ---------------------------------------------------------------------------
// Host orchestration
// ---------------------------------------------------------------------------
extern "C" void kernel_launch(void* const* d_in, const int* in_sizes, int n_in,
                              void* d_out, int out_size, void* d_ws, size_t ws_size,
                              hipStream_t stream) {
  (void)in_sizes; (void)n_in; (void)out_size; (void)ws_size;

  const float* xc  = (const float*)d_in[0];
  const float* xti = (const float*)d_in[1];
  const float* lat = (const float*)d_in[2];

  // params flattened: name in {ctoq,mhsa,mhca,qtot}, layer in {0,1},
  // leaves: Wq,Wk,Wv,Wo,W1,W2,ln1_g,ln1_b,ln2_g,ln2_b
  auto P = [&](int name, int layer, int leaf) -> const float* {
    return (const float*)d_in[3 + (name * 2 + layer) * 10 + leaf];
  };

  // ---- workspace carving ----
  char*  ws  = (char*)d_ws;
  size_t off = 0;
  auto alloc = [&](size_t bytes) -> void* {
    off = (off + 255) & ~(size_t)255;
    void* p = ws + off;
    off += bytes;
    return p;
  };
  const size_t DICT_ELEMS = 4 * (size_t)DM * DM + 2 * (size_t)DM * DFF; // 786432
  bf16*  wt  = (bf16*)alloc(8 * DICT_ELEMS * sizeof(bf16));   // 12.6 MB
  float* xq  = (float*)alloc((size_t)MB * LQ * DM * 4);       // 4 MB
  const size_t RMAX = (size_t)MB * NT;                        // 65536 rows
  bf16* aq  = (bf16*)alloc(RMAX * DM  * 2);
  bf16* ak  = (bf16*)alloc(RMAX * DM  * 2);
  bf16* qb  = (bf16*)alloc(RMAX * DM  * 2);
  bf16* kb  = (bf16*)alloc(RMAX * DM  * 2);
  bf16* vtb = (bf16*)alloc(RMAX * DM  * 2);   // per-head transposed V
  bf16* att = (bf16*)alloc(RMAX * DM  * 2);
  bf16* h2  = (bf16*)alloc(RMAX * DM  * 2);
  bf16* gb  = (bf16*)alloc(RMAX * DFF * 2);

  // ---- xt working copy lives in d_out (updated in place) ----
  hipMemcpyAsync(d_out, xti, (size_t)MB * NT * DM * sizeof(float),
                 hipMemcpyDeviceToDevice, stream);

  // ---- broadcast latents ----
  bcast_kernel<<<(MB * LQ * DM) / 256, 256, 0, stream>>>(lat, xq);

  // ---- transpose+convert all weights to bf16 W^T ----
  auto dictWT = [&](int name, int layer) -> bf16* {
    return wt + (size_t)(name * 2 + layer) * DICT_ELEMS;
  };
  for (int name = 0; name < 4; name++) {
    for (int layer = 0; layer < NLAYERS; layer++) {
      bf16* base = dictWT(name, layer);
      for (int wI = 0; wI < 4; wI++) {   // Wq,Wk,Wv,Wo : 256x256
        wprep_kernel<<<(DM * DM + 255) / 256, 256, 0, stream>>>(
            P(name, layer, wI), base + (size_t)wI * DM * DM, DM, DM);
      }
      // W1: 256x1024 -> Wt 1024x256
      wprep_kernel<<<(DM * DFF + 255) / 256, 256, 0, stream>>>(
          P(name, layer, 4), base + 4 * (size_t)DM * DM, DM, DFF);
      // W2: 1024x256 -> Wt 256x1024
      wprep_kernel<<<(DM * DFF + 255) / 256, 256, 0, stream>>>(
          P(name, layer, 5), base + 4 * (size_t)DM * DM + (size_t)DM * DFF, DFF, DM);
    }
  }

  auto gemm = [&](int op, const bf16* A, const bf16* Wt, const float* Cin,
                  float* Cf, bf16* Cb, int M, int N, int K, int nkshift,
                  float cscale) {
    dim3 g(N / 128, M / 128);
    if (op == 0)      gemm_kernel<0><<<g, 256, 0, stream>>>(A, Wt, Cin, Cf, Cb, M, N, K, nkshift, cscale);
    else if (op == 1) gemm_kernel<1><<<g, 256, 0, stream>>>(A, Wt, Cin, Cf, Cb, M, N, K, nkshift, cscale);
    else if (op == 2) gemm_kernel<2><<<g, 256, 0, stream>>>(A, Wt, Cin, Cf, Cb, M, N, K, nkshift, cscale);
    else              gemm_kernel<3><<<g, 256, 0, stream>>>(A, Wt, Cin, Cf, Cb, M, N, K, nkshift, cscale);
  };

  const float QSCALE = 0.17677669529663687f;   // 1/sqrt(DH)

  auto attn_block = [&](int name, int layer, float* Xq, const float* Xkv,
                        int rowsq, int rowskv, int nq, int nk, int nkshift,
                        bool self) {
    const float* ln1g = P(name, layer, 6);
    const float* ln1b = P(name, layer, 7);
    const float* ln2g = P(name, layer, 8);
    const float* ln2b = P(name, layer, 9);
    bf16* base = dictWT(name, layer);
    bf16* wqT = base;
    bf16* wkT = base + 1 * (size_t)DM * DM;
    bf16* wvT = base + 2 * (size_t)DM * DM;
    bf16* woT = base + 3 * (size_t)DM * DM;
    bf16* w1T = base + 4 * (size_t)DM * DM;
    bf16* w2T = base + 4 * (size_t)DM * DM + (size_t)DM * DFF;

    ln_kernel<<<rowsq / 8, 256, 0, stream>>>(Xq, ln1g, ln1b, aq);
    const bf16* akv = aq;
    if (!self) {
      ln_kernel<<<rowskv / 8, 256, 0, stream>>>(Xkv, ln1g, ln1b, ak);
      akv = ak;
    }
    gemm(0, aq,  wqT, nullptr, nullptr, qb,  rowsq,  DM, DM, 0, QSCALE);
    gemm(0, akv, wkT, nullptr, nullptr, kb,  rowskv, DM, DM, 0, 1.0f);
    gemm(3, akv, wvT, nullptr, nullptr, vtb, rowskv, DM, DM, nkshift, 1.0f);

    attn_kernel<<<(MB * NH * (nq / 32)) / 8, 256, 0, stream>>>(qb, kb, vtb, att, nq, nk);

    gemm(2, att, woT, Xq, Xq, nullptr, rowsq, DM, DM, 0, 1.0f);       // residual
    ln_kernel<<<rowsq / 8, 256, 0, stream>>>(Xq, ln2g, ln2b, h2);
    gemm(1, h2, w1T, nullptr, nullptr, gb, rowsq, DFF, DM, 0, 1.0f);  // gelu
    gemm(2, gb, w2T, Xq, Xq, nullptr, rowsq, DM, DFF, 0, 1.0f);       // residual
  };

  float* xt = (float*)d_out;
  for (int layer = 0; layer < NLAYERS; layer++) {
    attn_block(0, layer, xq, xc, MB * LQ, MB * NC, LQ, NC, 11, false);  // ctoq
    attn_block(1, layer, xq, xq, MB * LQ, MB * LQ, LQ, LQ, 7, true);    // mhsa
    attn_block(2, layer, xq, xq, MB * LQ, MB * LQ, LQ, LQ, 7, true);    // mhca
    attn_block(3, layer, xt, xq, MB * NT, MB * LQ, NT, LQ, 7, false);   // qtot
  }
}
